// GCN_55070070670247
// MI455X (gfx1250) — compile-verified, matrix operand order checked by reading, not run on hardware
//
#include <hip/hip_runtime.h>
#include <hip/hip_bf16.h>

// ---------------- types ----------------
typedef __attribute__((ext_vector_type(16))) __bf16 v16bf;
typedef __attribute__((ext_vector_type(8)))  float  v8f;
typedef __attribute__((ext_vector_type(4)))  float  v4f;
typedef __attribute__((ext_vector_type(4)))  unsigned short v4u16;

union ABu { v16bf v; v4f f4[2]; };

__device__ __forceinline__ unsigned short f2bf(float f) {
    unsigned int u = __float_as_uint(f);
    unsigned int r = u + 0x7FFFu + ((u >> 16) & 1u);   // round-to-nearest-even
    return (unsigned short)(r >> 16);
}

// ---------------- small elementwise kernels ----------------
__global__ void k_degree(const int* __restrict__ dst, float* __restrict__ deg, int E) {
    int e = blockIdx.x * blockDim.x + threadIdx.x;
    if (e < E) atomicAdd(&deg[dst[e]], 1.0f);
}

__global__ void k_dinv(const float* __restrict__ deg, float* __restrict__ dinv, int n) {
    int i = blockIdx.x * blockDim.x + threadIdx.x;
    if (i < n) dinv[i] = rsqrtf(fmaxf(deg[i], 1.0f));
}

// cast fp32 -> bf16, 4 elements per thread (b128 load / b64 store)
__global__ void k_cast4(const float* __restrict__ in, unsigned short* __restrict__ out, int n4) {
    int i = blockIdx.x * blockDim.x + threadIdx.x;
    if (i >= n4) return;
    v4f v = *(const v4f*)(in + (size_t)i * 4);
    v4u16 o;
    o.x = f2bf(v.x); o.y = f2bf(v.y); o.z = f2bf(v.z); o.w = f2bf(v.w);
    *(v4u16*)(out + (size_t)i * 4) = o;
}

// W [K, Ncols] fp32 (row-major) -> WT [Ncols, K] bf16
__global__ void k_transpose_cast(const float* __restrict__ W, unsigned short* __restrict__ WT,
                                 int K, int Ncols) {
    int idx = blockIdx.x * blockDim.x + threadIdx.x;
    if (idx >= K * Ncols) return;
    int k = idx / Ncols, n = idx % Ncols;
    WT[(size_t)n * K + k] = f2bf(W[idx]);
}

// out[i] = relu(agg[i] + bias[i % F]) cast to bf16
__global__ void k_bias_relu_cast(const float* __restrict__ agg, const float* __restrict__ bias,
                                 unsigned short* __restrict__ out, long long total, int F) {
    long long i = (long long)blockIdx.x * blockDim.x + threadIdx.x;
    if (i >= total) return;
    float v = agg[i] + bias[(int)(i & (F - 1))];
    out[i] = f2bf(fmaxf(v, 0.0f));
}

// ---------------- WMMA GEMM: C[M,NT*16] = A[M,128](bf16) * BT[NT*16,128](bf16)^T -------------
// Block = NT waves; the block owns one 16-row M strip. The A strip (16x128 bf16, 4KB) is
// staged in LDS once (padded to 136 shorts/row -> 272B stride, conflict-free half-wave reads),
// then each wave computes one 16x16 N tile: A from ds_load_b128, B from global (L2-resident).
// 16-bit A/B per-lane layout (ISA 7.12.2): lane&15 = row, lane>>4 selects K groups
// {0..7,16..23} (half=0) vs {8..15,24..31} (half=1); each group of 8 bf16 = one b128 access.
#define GEMM_K 128
#define LDS_PITCH 136

template <int NT>
__global__ void k_gemm_wmma_bf16(const unsigned short* __restrict__ A,
                                 const unsigned short* __restrict__ BT,
                                 float* __restrict__ C, int Ncols) {
    __shared__ unsigned short As[16][LDS_PITCH];

    const int tid  = threadIdx.x;
    const int lane = tid & 31;
    const int wave = tid >> 5;
    const int half = lane >> 4;      // 0/1
    const int l15  = lane & 15;
    const int m0 = blockIdx.y << 4;
    const int n0 = (blockIdx.x * NT + wave) << 4;

    // cooperative stage of the 16x128 A strip: 256 chunks of 8 bf16 (16B each)
    const int nchunk = 16 * (GEMM_K / 8);
    for (int c = tid; c < nchunk; c += NT * 32) {
        int row = c >> 4;            // GEMM_K/8 == 16 chunks per row
        int col = (c & 15) * 8;
        *(v4f*)&As[row][col] = *(const v4f*)(A + (size_t)(m0 + row) * GEMM_K + col);
    }
    __syncthreads();

    const unsigned short* brow = BT + (size_t)(n0 + l15) * GEMM_K;

    v8f acc = {};
    #pragma unroll
    for (int kk = 0; kk < GEMM_K; kk += 32) {
        ABu a, b;
        a.f4[0] = *(const v4f*)&As[l15][kk + half * 8];        // K = kk + half*8 + 0..7
        a.f4[1] = *(const v4f*)&As[l15][kk + 16 + half * 8];   // K = kk+16 + half*8 + 0..7
        const v4f* bp = (const v4f*)(brow + kk + half * 8);
        b.f4[0] = bp[0];
        b.f4[1] = bp[2];
        if (kk + 32 < GEMM_K)                                  // WGP-scope prefetch of next chunk
            __builtin_prefetch(brow + kk + 32, 0, 3);
        acc = __builtin_amdgcn_wmma_f32_16x16x32_bf16(
            false, a.v, false, b.v, (short)0, acc, false, false);
    }
    // C/D layout: VGPR r -> row m0 + r + half*8, col n0 + (lane&15)
    float* crow = C + (size_t)(m0 + half * 8) * Ncols + n0 + l15;
    #pragma unroll
    for (int r = 0; r < 8; ++r)
        crow[(size_t)r * Ncols] = acc[r];
}

// ---------------- edge aggregation: out[dst] += h[src] * dinv[src]*dinv[dst] ----------------
// (F/4) lanes per edge, each lane handles 4 consecutive features (b128 load, 4 f32 atomics).
__global__ void k_aggregate(const float* __restrict__ h, const int* __restrict__ src,
                            const int* __restrict__ dst, const float* __restrict__ dinv,
                            float* __restrict__ out, int E, int F, int shift, int mask) {
    long long tid = (long long)blockIdx.x * blockDim.x + threadIdx.x;
    long long e = tid >> shift;
    if (e >= E) return;
    int f4 = ((int)tid & mask) * 4;
    int s = src[e], d = dst[e];
    float w = dinv[s] * dinv[d];
    v4f v = *(const v4f*)(h + (size_t)s * F + f4);
    float* o = out + (size_t)d * F + f4;
    atomicAdd(o + 0, v.x * w);
    atomicAdd(o + 1, v.y * w);
    atomicAdd(o + 2, v.z * w);
    atomicAdd(o + 3, v.w * w);
}

// ---------------- per-node log_softmax over 16 classes ----------------
__global__ void k_logsoftmax16(const float* __restrict__ agg, const float* __restrict__ bias,
                               float* __restrict__ out, int n) {
    int i = blockIdx.x * blockDim.x + threadIdx.x;
    if (i >= n) return;
    float v[16];
    float mx = -3.402823466e+38f;
    #pragma unroll
    for (int c = 0; c < 16; ++c) {
        v[c] = agg[(size_t)i * 16 + c] + bias[c];
        mx = fmaxf(mx, v[c]);
    }
    float sum = 0.0f;
    #pragma unroll
    for (int c = 0; c < 16; ++c) sum += __expf(v[c] - mx);
    float lse = mx + __logf(sum);
    #pragma unroll
    for (int c = 0; c < 16; ++c)
        out[(size_t)i * 16 + c] = v[c] - lse;
}

// ---------------- host launch ----------------
extern "C" void kernel_launch(void* const* d_in, const int* in_sizes, int n_in,
                              void* d_out, int out_size, void* d_ws, size_t ws_size,
                              hipStream_t stream) {
    (void)n_in; (void)out_size; (void)ws_size;
    const float* x  = (const float*)d_in[0];
    const int*   ei = (const int*)  d_in[1];
    const float* W1 = (const float*)d_in[2];
    const float* b1 = (const float*)d_in[3];
    const float* W2 = (const float*)d_in[4];
    const float* b2 = (const float*)d_in[5];

    const int N = 50000, Fin = 128, H = 128, Cc = 16;
    const int E = in_sizes[1] / 2;
    const int* src = ei;
    const int* dst = ei + E;

    char* ws = (char*)d_ws;
    size_t off = 0;
    auto alloc = [&](size_t bytes) {
        size_t o = off;
        off = (off + bytes + 255) & ~(size_t)255;
        return o;
    };
    float*          deg  = (float*)         (ws + alloc((size_t)N * 4));
    float*          dinv = (float*)         (ws + alloc((size_t)N * 4));
    unsigned short* xb   = (unsigned short*)(ws + alloc((size_t)N * Fin * 2)); // reused as h1b
    unsigned short* w1t  = (unsigned short*)(ws + alloc((size_t)H * Fin * 2));
    unsigned short* w2t  = (unsigned short*)(ws + alloc((size_t)Cc * H * 2));
    float*          h1   = (float*)         (ws + alloc((size_t)N * H * 4));   // reused as h2
    float*          agg1 = (float*)         (ws + alloc((size_t)N * H * 4));   // head reused as agg2

    // init
    hipMemsetAsync(deg, 0, (size_t)N * 4, stream);
    hipMemsetAsync(agg1, 0, (size_t)N * H * 4, stream);

    // degrees and symmetric norm
    k_degree<<<(E + 255) / 256, 256, 0, stream>>>(dst, deg, E);
    k_dinv  <<<(N + 255) / 256, 256, 0, stream>>>(deg, dinv, N);

    // bf16 operands
    int n4 = N * Fin / 4;
    k_cast4<<<(n4 + 255) / 256, 256, 0, stream>>>(x, xb, n4);
    k_transpose_cast<<<(Fin * H + 255) / 256, 256, 0, stream>>>(W1, w1t, Fin, H);
    k_transpose_cast<<<(H * Cc + 255) / 256, 256, 0, stream>>>(W2, w2t, H, Cc);

    // layer 1: H1 = X @ W1  (8 waves/block, A strip staged in LDS, WMMA bf16 -> fp32)
    k_gemm_wmma_bf16<8><<<dim3(1, N / 16), 8 * 32, 0, stream>>>(xb, w1t, h1, H);

    // layer 1 aggregate: 32 lanes per edge (F=128 -> shift 5, mask 31)
    {
        long long total = (long long)E * (H / 4);
        k_aggregate<<<(int)((total + 255) / 256), 256, 0, stream>>>(
            h1, src, dst, dinv, agg1, E, H, 5, 31);
    }

    // bias + relu + cast to bf16 (into xb, X no longer needed)
    {
        long long total = (long long)N * H;
        k_bias_relu_cast<<<(int)((total + 255) / 256), 256, 0, stream>>>(
            agg1, b1, xb, total, H);
    }

    // zero agg2 (reuses head of agg1; agg1 fully consumed above)
    hipMemsetAsync(agg1, 0, (size_t)N * Cc * 4, stream);

    // layer 2: H2 = relu(...) @ W2  (1 wave/block; h2 reuses h1 buffer)
    k_gemm_wmma_bf16<1><<<dim3(1, N / 16), 32, 0, stream>>>(xb, w2t, h1, Cc);

    // layer 2 aggregate: 4 lanes per edge (F=16 -> shift 2, mask 3)
    {
        long long total = (long long)E * (Cc / 4);
        k_aggregate<<<(int)((total + 255) / 256), 256, 0, stream>>>(
            h1, src, dst, dinv, agg1, E, Cc, 2, 3);
    }

    // bias + log_softmax -> output
    k_logsoftmax16<<<(N + 255) / 256, 256, 0, stream>>>(agg1, b2, (float*)d_out, N);
}
